// ExampleNet_83047487636112
// MI455X (gfx1250) — compile-verified, matrix operand order checked by reading, not run on hardware
//
#include <hip/hip_runtime.h>
#include <hip/hip_bf16.h>

typedef __attribute__((ext_vector_type(16))) _Float16 v16h;
typedef __attribute__((ext_vector_type(8)))  float    v8f;
typedef unsigned int u32;
typedef unsigned char u8;

#define BB 4
#define HH 256
#define WW 256
#define HT 513   // convT output spatial
#define HO 511   // final output spatial

union F16x16 { v16h v; u32 u[8]; };
union PK { _Float16 h[2]; u32 u; };

// ---------------- utility kernels ----------------

__global__ void zero_u32_kernel(u32* p, long long n) {
    long long i = (long long)blockIdx.x * blockDim.x + threadIdx.x;
    long long stride = (long long)gridDim.x * blockDim.x;
    for (; i < n; i += stride) p[i] = 0u;
}

// one thread per (point, channel-pair): atomic scatter-add into dense fp32 grid
__global__ void scatter_kernel(const float* __restrict__ feat,
                               const int* __restrict__ coors,
                               float* __restrict__ x, u8* __restrict__ mask0, int npts) {
    int i = blockIdx.x * blockDim.x + threadIdx.x;
    int pt = i >> 4, cp = i & 15;
    if (pt >= npts) return;
    int b = coors[pt * 3 + 0];
    int y = coors[pt * 3 + 1];
    int xx = coors[pt * 3 + 2];
    long long pix = ((long long)b * HH + y) * WW + xx;
    long long base = pix * 32 + cp * 2;
    atomicAdd(&x[base + 0], feat[pt * 32 + cp * 2 + 0]);
    atomicAdd(&x[base + 1], feat[pt * 32 + cp * 2 + 1]);
    if (cp == 0) mask0[pix] = (u8)1;
}

// mask1 = (3x3 SAME dilation of mask0) > 0
__global__ void dilate_mask_kernel(const u8* __restrict__ m0, u8* __restrict__ m1) {
    int i = blockIdx.x * blockDim.x + threadIdx.x;
    if (i >= BB * HH * WW) return;
    int x = i % WW, y = (i / WW) % HH, b = i / (WW * HH);
    u8 acc = 0;
    for (int dy = -1; dy <= 1; ++dy) {
        int yy = y + dy; if (yy < 0 || yy >= HH) continue;
        for (int dx = -1; dx <= 1; ++dx) {
            int xx = x + dx; if (xx < 0 || xx >= WW) continue;
            acc |= m0[((long long)b * HH + yy) * WW + xx];
        }
    }
    m1[i] = acc ? (u8)1 : (u8)0;
}

// mask4 = (conv_transpose(mask1, ones3, stride 2) > 0) over (513,513)
__global__ void mask4_kernel(const u8* __restrict__ m1, u8* __restrict__ m4) {
    int i = blockIdx.x * blockDim.x + threadIdx.x;
    if (i >= BB * HT * HT) return;
    int X = i % HT, Y = (i / HT) % HT, b = i / (HT * HT);
    u8 acc = 0;
    for (int ky = 0; ky < 3; ++ky) {
        int u = Y + ky - 2;
        if (u < 0 || (u & 1)) continue;
        int y = u >> 1; if (y >= HH) continue;
        for (int kx = 0; kx < 3; ++kx) {
            int v = X + kx - 2;
            if (v < 0 || (v & 1)) continue;
            int x = v >> 1; if (x >= WW) continue;
            acc |= m1[((long long)b * HH + y) * WW + x];
        }
    }
    m4[i] = acc ? (u8)1 : (u8)0;
}

// Repack fp32 weights (3,3,CIN,64) into f16 WMMA B-fragment order:
// flat index = (((tap*KS + ks)*4 + ntile)*32 + lane)*8 + v   (dwords, 2 f16 each)
// lane: N = ntile*16 + (lane&15), half = lane>>4
// v:    K = ks*32 + ((v&4)?16:0) + (v&3)*2 + half*8   (packed K, K+1)
__global__ void swizzle_w_kernel(const float* __restrict__ w, u32* __restrict__ wsw, int CIN) {
    int KS = CIN / 32;
    int total = 9 * KS * 4 * 32 * 8;
    int d = blockIdx.x * blockDim.x + threadIdx.x;
    if (d >= total) return;
    int v = d & 7, lane = (d >> 3) & 31, nt = (d >> 8) & 3;
    int rest = d >> 10;
    int ks = rest % KS, t = rest / KS;
    int half = lane >> 4, n = lane & 15;
    int N = nt * 16 + n;
    int K = ks * 32 + ((v & 4) ? 16 : 0) + (v & 3) * 2 + half * 8;
    PK p;
    p.h[0] = (_Float16)w[((long long)t * CIN + K) * 64 + N];
    p.h[1] = (_Float16)w[((long long)t * CIN + K + 1) * 64 + N];
    wsw[d] = p.u;
}

// ---------------- WMMA conv core ----------------
// Each wave computes 32 pixels x 64 channels: 2 M-tiles x 4 N-tiles = 8 accumulators.
// Each B fragment (global, L2-resident) is reused by two WMMAs (both M-tiles).

template <int CIN>
__device__ __forceinline__ void conv_core(const _Float16* __restrict__ tile,
                                          const u32* __restrict__ wsw,
                                          v8f* acc, int wavePix, int lane) {
    const int KS = CIN / 32;
    const int half = lane >> 4, m = lane & 15;
    for (int t = 0; t < 9; ++t) {
        const int tr = t / 3, tc = t - tr * 3;
        const _Float16* lrow0 = tile + ((tr * 130) + wavePix + m + tc) * CIN;
        const _Float16* lrow1 = lrow0 + 16 * CIN;
        for (int ks = 0; ks < KS; ++ks) {
            F16x16 A0, A1;
#pragma unroll
            for (int v = 0; v < 8; ++v) {
                int K = ks * 32 + ((v & 4) ? 16 : 0) + (v & 3) * 2 + half * 8;
                A0.u[v] = *(const u32*)(lrow0 + K);
                A1.u[v] = *(const u32*)(lrow1 + K);
            }
            const u32* wbase = wsw + ((t * KS + ks) * 4) * 256;
#pragma unroll
            for (int nt = 0; nt < 4; ++nt) {
                F16x16 Bf;
                const uint4* wp = (const uint4*)(wbase + nt * 256 + lane * 8);
                uint4 q0 = wp[0], q1 = wp[1];
                Bf.u[0] = q0.x; Bf.u[1] = q0.y; Bf.u[2] = q0.z; Bf.u[3] = q0.w;
                Bf.u[4] = q1.x; Bf.u[5] = q1.y; Bf.u[6] = q1.z; Bf.u[7] = q1.w;
                acc[nt] = __builtin_amdgcn_wmma_f32_16x16x32_f16(
                    false, A0.v, false, Bf.v, (short)0, acc[nt], false, false);
                acc[4 + nt] = __builtin_amdgcn_wmma_f32_16x16x32_f16(
                    false, A1.v, false, Bf.v, (short)0, acc[4 + nt], false, false);
            }
        }
    }
}

// MODE 0: src fp32 (B,256,256,32), SAME pad, mask1, out f16 64ch
// MODE 1: src f16  (B,256,256,64), SAME pad, mask1, out f16 64ch
// MODE 2: src f16  (B,256,256,64), zero-dilated+pad2 staging (transposed conv),
//         mask4, out f16 (B,513,513,64)
// MODE 3: src f16  (B,513,513,64), VALID, no mask, out fp32 (B,511,511,64)
template <int CIN, int MODE>
__global__ void __launch_bounds__(128)
conv3x3_wmma_kernel(const void* __restrict__ src_, const u32* __restrict__ wsw,
                    const float* __restrict__ bias, const u8* __restrict__ mask,
                    void* __restrict__ out_) {
    constexpr int OUTH = (MODE == 2) ? HT : (MODE == 3 ? HO : HH);
    constexpr int OUTW = (MODE == 2) ? HT : (MODE == 3 ? HO : WW);
    constexpr bool MASKED = (MODE != 3);
    constexpr int SRCDW = CIN / 2;           // dwords per source pixel (f16 pairs)

    __shared__ _Float16 tile[3 * 130 * CIN];

    const int tid = threadIdx.x;
    const int xb = blockIdx.x * 128;
    const int b = blockIdx.y / OUTH;
    const int oy = blockIdx.y % OUTH;

    // ---- stage 3x130xCIN halo tile into LDS ----
    const int totaldw = 3 * 130 * SRCDW;
    for (int d = tid; d < totaldw; d += 128) {
        int cp = d % SRCDW;
        int rc = d / SRCDW;
        int row = rc / 130, col = rc % 130;
        u32 val = 0u;
        if (MODE == 0 || MODE == 1) {
            int y = oy - 1 + row, x = xb - 1 + col;
            if (y >= 0 && y < HH && x >= 0 && x < WW) {
                long long pix = ((long long)b * HH + y) * WW + x;
                if (MODE == 0) {
                    const float* s = (const float*)src_;
                    PK p;
                    p.h[0] = (_Float16)s[pix * CIN + cp * 2 + 0];
                    p.h[1] = (_Float16)s[pix * CIN + cp * 2 + 1];
                    val = p.u;
                } else {
                    val = ((const u32*)src_)[pix * SRCDW + cp];
                }
            }
        } else if (MODE == 2) {
            // padded-dilated coords: Dpad[p,q] = in[(p-2)/2,(q-2)/2] iff even & in range
            int pu = oy + row - 2, pv = xb + col - 2;
            if (pu >= 0 && pv >= 0 && !(pu & 1) && !(pv & 1)) {
                int y = pu >> 1, x = pv >> 1;
                if (y < HH && x < WW) {
                    long long pix = ((long long)b * HH + y) * WW + x;
                    val = ((const u32*)src_)[pix * SRCDW + cp];
                }
            }
        } else { // MODE 3, VALID over 513x513
            int y = oy + row, x = xb + col;
            if (y < HT && x < HT) {
                long long pix = ((long long)b * HT + y) * HT + x;
                val = ((const u32*)src_)[pix * SRCDW + cp];
            }
        }
        ((u32*)tile)[d] = val;
    }
    __syncthreads();

    // ---- per-wave WMMA compute: 32 pixels x 64 channels ----
    const int lane = tid & 31;
    const int wavePix = (tid >> 5) * 32;

    v8f z = {0.f, 0.f, 0.f, 0.f, 0.f, 0.f, 0.f, 0.f};
    v8f acc[8] = {z, z, z, z, z, z, z, z};
    conv_core<CIN>(tile, wsw, acc, wavePix, lane);

    // ---- epilogue: bias, mask, relu, store (C/D layout: N=lane&15, M=r+8*(lane>>4)) ----
    const int n = lane & 15, half = lane >> 4;
    float bn[4];
#pragma unroll
    for (int nt = 0; nt < 4; ++nt) bn[nt] = bias[nt * 16 + n];
    const u8* mrow = MASKED ? &mask[((long long)b * OUTH + oy) * OUTW] : (const u8*)nullptr;

#pragma unroll
    for (int mt = 0; mt < 2; ++mt) {
#pragma unroll
        for (int r = 0; r < 8; ++r) {
            int gx = xb + wavePix + mt * 16 + r + 8 * half;
            if (gx >= OUTW) continue;
            float mv = MASKED ? (float)mrow[gx] : 1.0f;
            long long ob = (((long long)b * OUTH + oy) * OUTW + gx) * 64;
#pragma unroll
            for (int nt = 0; nt < 4; ++nt) {
                float vL = (acc[mt * 4 + nt][r] + bn[nt]) * mv;
                vL = fmaxf(vL, 0.0f);
                if (MODE == 3)
                    ((float*)out_)[ob + nt * 16 + n] = vL;
                else
                    ((_Float16*)out_)[ob + nt * 16 + n] = (_Float16)vL;
            }
        }
    }
}

// ---------------- launcher ----------------

extern "C" void kernel_launch(void* const* d_in, const int* in_sizes, int n_in,
                              void* d_out, int out_size, void* d_ws, size_t ws_size,
                              hipStream_t stream) {
    const float* features = (const float*)d_in[0];
    const int*   coors    = (const int*)d_in[1];
    const float* w1 = (const float*)d_in[3];  const float* b1 = (const float*)d_in[4];
    const float* w2 = (const float*)d_in[5];  const float* b2 = (const float*)d_in[6];
    const float* w3 = (const float*)d_in[7];  const float* b3 = (const float*)d_in[8];
    const float* wt = (const float*)d_in[9];  const float* bt = (const float*)d_in[10];
    const float* w5 = (const float*)d_in[11]; const float* b5 = (const float*)d_in[12];
    const int npts = in_sizes[0] / 32;

    // workspace carve-up (256B aligned)
    char* ws = (char*)d_ws;
    size_t off = 0;
    auto carve = [&](size_t bytes) -> char* {
        char* p = ws + off;
        off = (off + bytes + 255) & ~(size_t)255;
        return p;
    };
    float*     x32   = (float*)    carve((size_t)BB * HH * WW * 32 * 4);
    u8*        mask0 = (u8*)       carve((size_t)BB * HH * WW);
    u8*        mask1 = (u8*)       carve((size_t)BB * HH * WW);
    u8*        mask4 = (u8*)       carve((size_t)BB * HT * HT);
    _Float16*  h1    = (_Float16*) carve((size_t)BB * HH * WW * 64 * 2);
    _Float16*  h2    = (_Float16*) carve((size_t)BB * HH * WW * 64 * 2);
    _Float16*  hT    = (_Float16*) carve((size_t)BB * HT * HT * 64 * 2);
    u32*       wsw1  = (u32*)      carve((size_t)9 * 1 * 4 * 32 * 8 * 4);
    u32*       wsw2  = (u32*)      carve((size_t)9 * 2 * 4 * 32 * 8 * 4);
    u32*       wsw3  = (u32*)      carve((size_t)9 * 2 * 4 * 32 * 8 * 4);
    u32*       wswt  = (u32*)      carve((size_t)9 * 2 * 4 * 32 * 8 * 4);
    u32*       wsw5  = (u32*)      carve((size_t)9 * 2 * 4 * 32 * 8 * 4);

    // zero accumulation grid + occupancy mask
    {
        long long nx = (long long)BB * HH * WW * 32;
        zero_u32_kernel<<<4096, 256, 0, stream>>>((u32*)x32, nx);
        long long nm = (long long)BB * HH * WW / 4;
        zero_u32_kernel<<<256, 256, 0, stream>>>((u32*)mask0, nm);
    }

    // scatter points
    scatter_kernel<<<(npts * 16 + 255) / 256, 256, 0, stream>>>(features, coors, x32, mask0, npts);

    // masks
    dilate_mask_kernel<<<(BB * HH * WW + 255) / 256, 256, 0, stream>>>(mask0, mask1);
    mask4_kernel<<<(BB * HT * HT + 255) / 256, 256, 0, stream>>>(mask1, mask4);

    // repack weights into WMMA fragment order (f16)
    swizzle_w_kernel<<<(9 * 1 * 4 * 32 * 8 + 255) / 256, 256, 0, stream>>>(w1, wsw1, 32);
    swizzle_w_kernel<<<(9 * 2 * 4 * 32 * 8 + 255) / 256, 256, 0, stream>>>(w2, wsw2, 64);
    swizzle_w_kernel<<<(9 * 2 * 4 * 32 * 8 + 255) / 256, 256, 0, stream>>>(w3, wsw3, 64);
    swizzle_w_kernel<<<(9 * 2 * 4 * 32 * 8 + 255) / 256, 256, 0, stream>>>(wt, wswt, 64);
    swizzle_w_kernel<<<(9 * 2 * 4 * 32 * 8 + 255) / 256, 256, 0, stream>>>(w5, wsw5, 64);

    // conv1: fp32 32ch -> f16 64ch, masked
    conv3x3_wmma_kernel<32, 0><<<dim3(2, BB * HH), 128, 0, stream>>>(x32, wsw1, b1, mask1, h1);
    // conv2, conv3: f16 64ch, masked
    conv3x3_wmma_kernel<64, 1><<<dim3(2, BB * HH), 128, 0, stream>>>(h1, wsw2, b2, mask1, h2);
    conv3x3_wmma_kernel<64, 1><<<dim3(2, BB * HH), 128, 0, stream>>>(h2, wsw3, b3, mask1, h1);
    // transposed conv as dense conv over zero-dilated input, masked by mask4
    conv3x3_wmma_kernel<64, 2><<<dim3(5, BB * HT), 128, 0, stream>>>(h1, wswt, bt, mask4, hT);
    // final VALID conv, fp32 output
    conv3x3_wmma_kernel<64, 3><<<dim3(4, BB * HO), 128, 0, stream>>>(hT, wsw5, b5, nullptr, (float*)d_out);
}
